// MatryoshkaNautilus_27084063768962
// MI455X (gfx1250) — matrix operand order repaired and checked by reading, not compiled, not run on hardware
//
#include <hip/hip_runtime.h>
#include <math.h>

// Problem constants (from reference)
#define BB   4
#define TT   1024
#define DD   768
#define NCH  7
#define BT   (BB * TT)          // 4096
#define NDD  (NCH * DD)         // 5376
#define STEP_C   4000.0f
#define WARMUP_C 2000.0f

typedef __attribute__((ext_vector_type(16))) __bf16 v16bf;
typedef __attribute__((ext_vector_type(8)))  float  v8f;

union Frag16 { unsigned int u[8]; v16bf v; };
static_assert(sizeof(Frag16) == 32, "frag size");

// CDNA5 async global->LDS copy path (ASYNCcnt), if the toolchain exposes it.
// Probe round confirmed the builtin's params are pointers to a 4 x i32 vector.
#if defined(__gfx1250__) && \
    __has_builtin(__builtin_amdgcn_global_load_async_to_lds_b128) && \
    __has_builtin(__builtin_amdgcn_s_wait_asynccnt)
#define HAVE_ASYNC_LDS 1
typedef int v4i __attribute__((vector_size(16)));
typedef __attribute__((address_space(1))) v4i as1_v4i;
typedef __attribute__((address_space(3))) v4i as3_v4i;
#else
#define HAVE_ASYNC_LDS 0
#endif

// ---------------------------------------------------------------------------
// Elementwise fp32 -> bf16 convert (merge weights)
// ---------------------------------------------------------------------------
__global__ __launch_bounds__(256)
void cvt_bf16_kernel(const float* __restrict__ in, __bf16* __restrict__ out, int n) {
  int i = blockIdx.x * 256 + threadIdx.x;
  if (i < n) out[i] = (__bf16)in[i];
}

// ---------------------------------------------------------------------------
// Tiled transpose (+ optional fp32->bf16 convert): out[c][r] = (bf16)in[r][c]
// in: R x C row-major, out: C x R row-major.  Batched via blockIdx.z.
// ---------------------------------------------------------------------------
template <typename IT>
__global__ __launch_bounds__(256)
void transpose_bf16_kernel(const IT* __restrict__ in, __bf16* __restrict__ out,
                           int R, int Ccols, long sIn, long sOut) {
  __shared__ __bf16 tile[32][33];
  in  += (long)blockIdx.z * sIn;
  out += (long)blockIdx.z * sOut;
  const int c0 = blockIdx.x * 32;
  const int r0 = blockIdx.y * 32;
  const int tid = threadIdx.x;
  const int tx = tid & 31;      // fast index
  const int ty = tid >> 5;      // 0..7
#pragma unroll
  for (int i = 0; i < 4; ++i) {
    int rl = ty + i * 8;        // row in tile
    tile[tx][rl] = (__bf16)in[(long)(r0 + rl) * Ccols + (c0 + tx)];
  }
  __syncthreads();
#pragma unroll
  for (int i = 0; i < 4; ++i) {
    int cl = ty + i * 8;        // out row (= input col)
    out[(long)(c0 + cl) * R + (r0 + tx)] = tile[cl][tx];
  }
}

// ---------------------------------------------------------------------------
// Pre-layernorm: h = LN(x)*g+b ; also bf16 copy, and zero the seq accumulator
// ---------------------------------------------------------------------------
__global__ __launch_bounds__(256)
void ln_pre_kernel(const float* __restrict__ x,
                   const float* __restrict__ g, const float* __restrict__ b,
                   float* __restrict__ h, __bf16* __restrict__ hb,
                   float* __restrict__ seq) {
  const int row = blockIdx.x;
  const float* xr = x + (long)row * DD;
  __shared__ float red[256];
  float s = 0.f;
  for (int j = threadIdx.x; j < DD; j += 256) s += xr[j];
  red[threadIdx.x] = s; __syncthreads();
  for (int off = 128; off > 0; off >>= 1) {
    if (threadIdx.x < off) red[threadIdx.x] += red[threadIdx.x + off];
    __syncthreads();
  }
  const float mean = red[0] * (1.0f / DD);
  __syncthreads();
  float var = 0.f;
  for (int j = threadIdx.x; j < DD; j += 256) { float d = xr[j] - mean; var += d * d; }
  red[threadIdx.x] = var; __syncthreads();
  for (int off = 128; off > 0; off >>= 1) {
    if (threadIdx.x < off) red[threadIdx.x] += red[threadIdx.x + off];
    __syncthreads();
  }
  const float rstd = rsqrtf(red[0] * (1.0f / DD) + 1e-5f);
  for (int j = threadIdx.x; j < DD; j += 256) {
    float val = (xr[j] - mean) * rstd * g[j] + b[j];
    h[(long)row * DD + j]  = val;
    hb[(long)row * DD + j] = (__bf16)val;
    seq[(long)row * DD + j] = 0.f;
  }
}

// ---------------------------------------------------------------------------
// inp = h + seq (fp32) and bf16 copy
// ---------------------------------------------------------------------------
__global__ __launch_bounds__(256)
void prep_input_kernel(const float* __restrict__ h, const float* __restrict__ seq,
                       float* __restrict__ inp, __bf16* __restrict__ inpb, int n) {
  int i = blockIdx.x * 256 + threadIdx.x;
  if (i < n) { float v = h[i] + seq[i]; inp[i] = v; inpb[i] = (__bf16)v; }
}

// ---------------------------------------------------------------------------
// NT WMMA GEMM, bf16 x bf16 -> CT (float or bf16), alpha scale.
//   C[M x Nn] = alpha * A[M x K] @ Bm[Nn x K]^T     (both K-major row storage)
// Block tile 128(M) x 64(N), BK=64, double-buffered LDS (48 KB), 8 waves in
// 4x2; each wave owns a 32x32 C block = 4 accumulators, 8 WMMAs per K-tile.
// Staging uses GLOBAL_LOAD_ASYNC_TO_LDS_B128 when available (ASYNCcnt),
// otherwise uint4 register staging.  Batched via gridDim.z.
// CAUSAL: skip blocks entirely above the diagonal (rows m read only cols <= m).
// ---------------------------------------------------------------------------
#define BKT 64
template <bool CAUSAL, typename CT>
__global__ __launch_bounds__(256)
void gemm_nt_wmma_kernel(const __bf16* __restrict__ A, const __bf16* __restrict__ Bm,
                         CT* __restrict__ C, int K,
                         int lda, int ldb, int ldc,
                         long sA, long sB, long sC, float alpha) {
  const int m0 = blockIdx.y * 128;
  const int n0 = blockIdx.x * 64;
  if (CAUSAL && n0 > m0 + 127) return;   // fully masked score block: never read

  __shared__ __bf16 As[2][128][BKT];
  __shared__ __bf16 Bs[2][64][BKT];
  const int bz = blockIdx.z;
  A  += (long)bz * sA;
  Bm += (long)bz * sB;
  C  += (long)bz * sC;

  const int tid  = threadIdx.x;
  const int lane = tid & 31;     // wave32
  const int wv   = tid >> 5;     // 8 waves
  const int wr   = wv >> 1;      // 0..3 : 32 M-rows each
  const int wc   = wv & 1;       // 0..1 : 32 N-cols each

  v8f acc[2][2] = {{{}, {}}, {{}, {}}};
  const int nk = K / BKT;

  // ---- stage one 128x64 A tile + 64x64 B tile (16B granules)
  auto stage = [&](int kt, int buf) {
    const int k0 = kt * BKT;
#if HAVE_ASYNC_LDS
#pragma unroll
    for (int j = 0; j < 4; ++j) {                 // A: 1024 x 16B, 4/thread
      int idx = tid + j * 256;
      int r   = idx >> 3;                         // 0..127
      int kc  = (idx & 7) << 3;                   // 0,8,..,56
      __builtin_amdgcn_global_load_async_to_lds_b128(
          (as1_v4i*)(A + (long)(m0 + r) * lda + k0 + kc),
          (as3_v4i*)(&As[buf][r][kc]), 0, 0);
    }
#pragma unroll
    for (int j = 0; j < 2; ++j) {                 // B: 512 x 16B, 2/thread
      int idx = tid + j * 256;
      int r   = idx >> 3;                         // 0..63
      int kc  = (idx & 7) << 3;
      __builtin_amdgcn_global_load_async_to_lds_b128(
          (as1_v4i*)(Bm + (long)(n0 + r) * ldb + k0 + kc),
          (as3_v4i*)(&Bs[buf][r][kc]), 0, 0);
    }
#else
    uint4 ra[4], rb[2];
#pragma unroll
    for (int j = 0; j < 4; ++j) {
      int idx = tid + j * 256;
      ra[j] = *(const uint4*)(A + (long)(m0 + (idx >> 3)) * lda + k0 + ((idx & 7) << 3));
    }
#pragma unroll
    for (int j = 0; j < 2; ++j) {
      int idx = tid + j * 256;
      rb[j] = *(const uint4*)(Bm + (long)(n0 + (idx >> 3)) * ldb + k0 + ((idx & 7) << 3));
    }
#pragma unroll
    for (int j = 0; j < 4; ++j) {
      int idx = tid + j * 256;
      *(uint4*)(&As[buf][idx >> 3][(idx & 7) << 3]) = ra[j];
    }
#pragma unroll
    for (int j = 0; j < 2; ++j) {
      int idx = tid + j * 256;
      *(uint4*)(&Bs[buf][idx >> 3][(idx & 7) << 3]) = rb[j];
    }
#endif
  };

  stage(0, 0);
  for (int kt = 0; kt < nk; ++kt) {
#if HAVE_ASYNC_LDS
    __builtin_amdgcn_s_wait_asynccnt(0);          // my LDS writes landed
#endif
    __syncthreads();                              // everyone's landed
    const int buf = kt & 1;
    if (kt + 1 < nk) stage(kt + 1, buf ^ 1);
    if (kt + 2 < nk) {                            // CDNA5 global_prefetch_b8
      const int kp = (kt + 2) * BKT;
      __builtin_prefetch(A + (long)(m0 + (tid >> 1)) * lda + kp, 0, 1);
      __builtin_prefetch(Bm + (long)(n0 + (tid >> 2)) * ldb + kp, 0, 1);
    }

    // ---- two 32-deep K sub-steps per tile
#pragma unroll
    for (int s = 0; s < 2; ++s) {
      const int ks = s * 32;
      Frag16 a0, a1, b0, b1;
      const int rbase = wr * 32 + (lane & 15);
      const int abase = ks + (lane >> 4) * 8;
#pragma unroll
      for (int j2 = 0; j2 < 4; ++j2) {
        a0.u[j2]     = *(const unsigned int*)(&As[buf][rbase][abase + 2 * j2]);
        a0.u[j2 + 4] = *(const unsigned int*)(&As[buf][rbase][abase + 16 + 2 * j2]);
        a1.u[j2]     = *(const unsigned int*)(&As[buf][rbase + 16][abase + 2 * j2]);
        a1.u[j2 + 4] = *(const unsigned int*)(&As[buf][rbase + 16][abase + 16 + 2 * j2]);
      }
      const int cbase = wc * 32 + (lane & 15);
      const int kbase = ks + (lane >> 4) * 8;
#pragma unroll
      for (int j2 = 0; j2 < 4; ++j2) {
        b0.u[j2]     = *(const unsigned int*)(&Bs[buf][cbase][kbase + 2 * j2]);
        b0.u[j2 + 4] = *(const unsigned int*)(&Bs[buf][cbase][kbase + 16 + 2 * j2]);
        b1.u[j2]     = *(const unsigned int*)(&Bs[buf][cbase + 16][kbase + 2 * j2]);
        b1.u[j2 + 4] = *(const unsigned int*)(&Bs[buf][cbase + 16][kbase + 16 + 2 * j2]);
      }
      acc[0][0] = __builtin_amdgcn_wmma_f32_16x16x32_bf16(false, a0.v, false, b0.v,
                                                          (short)0, acc[0][0], false, false);
      acc[0][1] = __builtin_amdgcn_wmma_f32_16x16x32_bf16(false, a0.v, false, b1.v,
                                                          (short)0, acc[0][1], false, false);
      acc[1][0] = __builtin_amdgcn_wmma_f32_16x16x32_bf16(false, a1.v, false, b0.v,
                                                          (short)0, acc[1][0], false, false);
      acc[1][1] = __builtin_amdgcn_wmma_f32_16x16x32_bf16(false, a1.v, false, b1.v,
                                                          (short)0, acc[1][1], false, false);
    }
  }

  // ---- epilogue (C layout: lane%16 = N col, VGPR v = M row (lane/16)*8+v)
#pragma unroll
  for (int mt = 0; mt < 2; ++mt) {
    const int mrow0 = m0 + wr * 32 + mt * 16 + (lane >> 4) * 8;
#pragma unroll
    for (int nt = 0; nt < 2; ++nt) {
      const int ncol = n0 + wc * 32 + nt * 16 + (lane & 15);
#pragma unroll
      for (int v2 = 0; v2 < 8; ++v2)
        C[(long)(mrow0 + v2) * ldc + ncol] = (CT)(acc[mt][nt][v2] * alpha);
    }
  }
}

// ---------------------------------------------------------------------------
// Causal softmax over S rows -> bf16 probabilities (zeros beyond diagonal)
// ---------------------------------------------------------------------------
__global__ __launch_bounds__(256)
void softmax_causal_kernel(const float* __restrict__ S, __bf16* __restrict__ P) {
  const int b = blockIdx.x / TT;
  const int t = blockIdx.x % TT;
  const float* srow = S + (long)b * TT * TT + (long)t * TT;
  __bf16* prow      = P + (long)b * TT * TT + (long)t * TT;
  const int valid = t + 1;
  __shared__ float red[256];
  float mx = -3.0e38f;
  for (int j = threadIdx.x; j < valid; j += 256) mx = fmaxf(mx, srow[j]);
  red[threadIdx.x] = mx; __syncthreads();
  for (int off = 128; off > 0; off >>= 1) {
    if (threadIdx.x < off) red[threadIdx.x] = fmaxf(red[threadIdx.x], red[threadIdx.x + off]);
    __syncthreads();
  }
  mx = red[0]; __syncthreads();
  float sum = 0.f;
  for (int j = threadIdx.x; j < valid; j += 256) sum += __expf(srow[j] - mx);
  red[threadIdx.x] = sum; __syncthreads();
  for (int off = 128; off > 0; off >>= 1) {
    if (threadIdx.x < off) red[threadIdx.x] += red[threadIdx.x + off];
    __syncthreads();
  }
  const float inv = 1.0f / red[0];
  for (int j = threadIdx.x; j < TT; j += 256)
    prow[j] = (__bf16)((j < valid) ? __expf(srow[j] - mx) * inv : 0.0f);
}

// ---------------------------------------------------------------------------
// Chamber epilogue: gate = sigmoid(inp.gw + gb); scale = softplus(sp);
//   val = scale*gate*mask*(O - inp);  seq += val  OR  deltas[:, i*D:] = bf16(val)
// ---------------------------------------------------------------------------
__global__ __launch_bounds__(256)
void chamber_epilogue_kernel(const float* __restrict__ O, const float* __restrict__ inp,
                             const float* __restrict__ gw, const float* __restrict__ gb,
                             const float* __restrict__ sp, float mask, int seqMode,
                             float* __restrict__ seq, __bf16* __restrict__ deltas,
                             int chamber) {
  const int row = blockIdx.x;
  const float* ir = inp + (long)row * DD;
  const float* orow = O + (long)row * DD;
  __shared__ float red[256];
  float s = 0.f;
  for (int j = threadIdx.x; j < DD; j += 256) s += ir[j] * gw[j];
  red[threadIdx.x] = s; __syncthreads();
  for (int off = 128; off > 0; off >>= 1) {
    if (threadIdx.x < off) red[threadIdx.x] += red[threadIdx.x + off];
    __syncthreads();
  }
  const float gate  = 1.0f / (1.0f + __expf(-(red[0] + gb[0])));
  const float spv   = sp[0];
  const float scale = (spv > 20.0f) ? spv : log1pf(__expf(spv));
  const float coef  = scale * gate * mask;
  if (seqMode) {
    for (int j = threadIdx.x; j < DD; j += 256)
      seq[(long)row * DD + j] += coef * (orow[j] - ir[j]);
  } else {
    for (int j = threadIdx.x; j < DD; j += 256)
      deltas[(long)row * NDD + (long)chamber * DD + j] = (__bf16)(coef * (orow[j] - ir[j]));
  }
}

// ---------------------------------------------------------------------------
// Final: e = (1-mg)*seq + mg*par ; e = LN(e) ; out = x + rg*e
// ---------------------------------------------------------------------------
__global__ __launch_bounds__(256)
void final_kernel(const float* __restrict__ x, const float* __restrict__ seq,
                  const float* __restrict__ par, const float* __restrict__ mode_logit,
                  const float* __restrict__ rgate, const float* __restrict__ g,
                  const float* __restrict__ b, float* __restrict__ out) {
  const int row = blockIdx.x;
  __shared__ float e[DD];
  __shared__ float red[256];
  const float mg = 1.0f / (1.0f + __expf(-mode_logit[0]));
  const float rg = rgate[0];
  float s = 0.f;
  for (int j = threadIdx.x; j < DD; j += 256) {
    float v = (1.0f - mg) * seq[(long)row * DD + j] + mg * par[(long)row * DD + j];
    e[j] = v; s += v;
  }
  red[threadIdx.x] = s; __syncthreads();
  for (int off = 128; off > 0; off >>= 1) {
    if (threadIdx.x < off) red[threadIdx.x] += red[threadIdx.x + off];
    __syncthreads();
  }
  const float mean = red[0] * (1.0f / DD);
  __syncthreads();
  float var = 0.f;
  for (int j = threadIdx.x; j < DD; j += 256) { float d = e[j] - mean; var += d * d; }
  red[threadIdx.x] = var; __syncthreads();
  for (int off = 128; off > 0; off >>= 1) {
    if (threadIdx.x < off) red[threadIdx.x] += red[threadIdx.x + off];
    __syncthreads();
  }
  const float rstd = rsqrtf(red[0] * (1.0f / DD) + 1e-5f);
  for (int j = threadIdx.x; j < DD; j += 256) {
    float ln = (e[j] - mean) * rstd * g[j] + b[j];
    out[(long)row * DD + j] = x[(long)row * DD + j] + rg * ln;
  }
}

// ---------------------------------------------------------------------------
// Host orchestration
// ---------------------------------------------------------------------------
static inline size_t alignUp(size_t v) { return (v + 255) & ~(size_t)255; }

extern "C" void kernel_launch(void* const* d_in, const int* in_sizes, int n_in,
                              void* d_out, int out_size, void* d_ws, size_t ws_size,
                              hipStream_t stream) {
  (void)in_sizes; (void)n_in; (void)out_size; (void)ws_size;
  const float* x        = (const float*)d_in[0];
  const float* Wq       = (const float*)d_in[1];
  const float* Wk       = (const float*)d_in[2];
  const float* gate_w   = (const float*)d_in[3];
  const float* gate_b   = (const float*)d_in[4];
  const float* scale_p  = (const float*)d_in[5];
  const float* merge_W  = (const float*)d_in[6];
  const float* mode_lg  = (const float*)d_in[7];
  const float* res_gate = (const float*)d_in[8];
  const float* lnp_g    = (const float*)d_in[9];
  const float* lnp_b    = (const float*)d_in[10];
  const float* lns_g    = (const float*)d_in[11];
  const float* lns_b    = (const float*)d_in[12];
  float* out = (float*)d_out;

  // workspace layout
  char* ws = (char*)d_ws;
  size_t off = 0;
  auto take = [&](size_t bytes) { char* p = ws + off; off += alignUp(bytes); return p; };
  float*  h     = (float*) take((size_t)BT * DD * 4);
  __bf16* hb    = (__bf16*)take((size_t)BT * DD * 2);
  __bf16* hbT   = (__bf16*)take((size_t)BT * DD * 2);   // per-batch D x T
  float*  seq   = (float*) take((size_t)BT * DD * 4);
  float*  inp   = (float*) take((size_t)BT * DD * 4);
  __bf16* inpb  = (__bf16*)take((size_t)BT * DD * 2);
  __bf16* inpbT = (__bf16*)take((size_t)BT * DD * 2);   // per-batch D x T
  __bf16* qb    = (__bf16*)take((size_t)BT * DD * 2);
  __bf16* kb    = (__bf16*)take((size_t)BT * DD * 2);
  float*  S     = (float*) take((size_t)BB * TT * TT * 4);
  __bf16* P     = (__bf16*)take((size_t)BB * TT * TT * 2);
  float*  O     = (float*) take((size_t)BT * DD * 4);
  __bf16* deltas= (__bf16*)take((size_t)BT * NDD * 2);
  float*  par   = (float*) take((size_t)BT * DD * 4);
  __bf16* WqbT  = (__bf16*)take((size_t)NCH * DD * DD * 2);  // per-chamber W^T (n,k)
  __bf16* WkbT  = (__bf16*)take((size_t)NCH * DD * DD * 2);
  __bf16* mwb   = (__bf16*)take((size_t)DD * NDD * 2);       // already (n,k)

  // curriculum masks (compile-time constants of the reference)
  float masks[NCH];
  {
    const float slope = 8.0f * NCH / WARMUP_C;
    for (int i = 0; i < NCH; ++i)
      masks[i] = 1.0f / (1.0f + expf(-slope * (STEP_C - WARMUP_C * (i + 0.5f) / NCH)));
  }
  const float qscale = 1.0f / sqrtf((float)DD);

  // weights: W^T bf16 (K-major for NT GEMM), merge_W plain bf16 (already n,k)
  {
    dim3 gW(DD / 32, DD / 32, NCH);
    transpose_bf16_kernel<float><<<gW, 256, 0, stream>>>(
        Wq, WqbT, DD, DD, (long)DD * DD, (long)DD * DD);
    transpose_bf16_kernel<float><<<gW, 256, 0, stream>>>(
        Wk, WkbT, DD, DD, (long)DD * DD, (long)DD * DD);
    int nm = DD * NDD;
    cvt_bf16_kernel<<<(nm + 255) / 256, 256, 0, stream>>>(merge_W, mwb, nm);
  }

  // pre-LN (also zeros seq) + hbT
  ln_pre_kernel<<<BT, 256, 0, stream>>>(x, lnp_g, lnp_b, h, hb, seq);
  {
    dim3 gT(DD / 32, TT / 32, BB);
    transpose_bf16_kernel<__bf16><<<gT, 256, 0, stream>>>(
        hb, hbT, TT, DD, (long)TT * DD, (long)TT * DD);
  }

  const dim3 gProj(DD / 64, BT / 128, 1);       // q/k projections: 4096x768x768
  const dim3 gScore(TT / 64, TT / 128, BB);     // scores: (T x T x D) per batch
  const dim3 gApply(DD / 64, TT / 128, BB);     // apply:  (T x D x T) per batch
  const dim3 gMerge(DD / 64, BT / 128, 1);      // merge:  4096x768x5376

  auto run_chamber = [&](int i, const __bf16* Ab, const __bf16* AbT,
                         const float* Af, int seqMode) {
    gemm_nt_wmma_kernel<false, __bf16><<<gProj, 256, 0, stream>>>(
        Ab, WqbT + (size_t)i * DD * DD, qb, DD, DD, DD, DD, 0, 0, 0, qscale);
    gemm_nt_wmma_kernel<false, __bf16><<<gProj, 256, 0, stream>>>(
        Ab, WkbT + (size_t)i * DD * DD, kb, DD, DD, DD, DD, 0, 0, 0, 1.0f);
    gemm_nt_wmma_kernel<true, float><<<gScore, 256, 0, stream>>>(
        qb, kb, S, DD, DD, DD, TT,
        (long)TT * DD, (long)TT * DD, (long)TT * TT, 1.0f);
    softmax_causal_kernel<<<BB * TT, 256, 0, stream>>>(S, P);
    gemm_nt_wmma_kernel<false, float><<<gApply, 256, 0, stream>>>(
        P, AbT, O, TT, TT, TT, DD,
        (long)TT * TT, (long)TT * DD, (long)TT * DD, 1.0f);
    chamber_epilogue_kernel<<<BT, 256, 0, stream>>>(
        O, Af, gate_w + (size_t)i * DD, gate_b + i, scale_p + i,
        masks[i], seqMode, seq, deltas, i);
  };

  // sequential chain: inp = h + seq before each chamber, plus V^T for apply GEMM
  for (int i = 0; i < NCH; ++i) {
    int n = BT * DD;
    prep_input_kernel<<<(n + 255) / 256, 256, 0, stream>>>(h, seq, inp, inpb, n);
    dim3 gT(DD / 32, TT / 32, BB);
    transpose_bf16_kernel<__bf16><<<gT, 256, 0, stream>>>(
        inpb, inpbT, TT, DD, (long)TT * DD, (long)TT * DD);
    run_chamber(i, inpb, inpbT, inp, /*seqMode=*/1);
  }
  // parallel deltas on h
  for (int i = 0; i < NCH; ++i)
    run_chamber(i, hb, hbT, h, /*seqMode=*/0);

  // par = deltas @ merge_W^T   (merge_W rows are already K-major)
  gemm_nt_wmma_kernel<false, float><<<gMerge, 256, 0, stream>>>(
      deltas, mwb, par, NDD, NDD, NDD, DD, 0, 0, 0, 1.0f);

  // blend + post-LN + residual
  final_kernel<<<BT, 256, 0, stream>>>(x, seq, par, mode_lg, res_gate, lns_g, lns_b, out);
}
